// MinkUNetBase_78958678769868
// MI455X (gfx1250) — compile-verified, matrix operand order checked by reading, not run on hardware
//
#include <hip/hip_runtime.h>
#include <hip/hip_bf16.h>
#include <stdint.h>
#include <stddef.h>

// ---------------------------------------------------------------------------
// Types for CDNA5 WMMA (wave32): v_wmma_f32_16x16x32_bf16
// ---------------------------------------------------------------------------
typedef __bf16 bf16;
typedef __attribute__((ext_vector_type(16))) __bf16 v16bf;
typedef __attribute__((ext_vector_type(8)))  float  v8f;

#define MAXOFF 27   // largest WMMA-path offset count (3^3 kernels)

struct ConvMaps {
  const int* ii[MAXOFF];   // ii[k]==nullptr => identity (dense 1x1 conv)
  const int* oi[MAXOFF];
  int        len[MAXOFF];
};

// ---------------------------------------------------------------------------
// Utility kernels
// ---------------------------------------------------------------------------
__global__ void fillz_kernel(float* p, long long n) {
  for (long long i = (long long)blockIdx.x * blockDim.x + threadIdx.x; i < n;
       i += (long long)gridDim.x * blockDim.x)
    p[i] = 0.0f;
}

__global__ void cast_bf16_kernel(const float* __restrict__ x, bf16* __restrict__ y,
                                 long long n) {
  for (long long i = (long long)blockIdx.x * blockDim.x + threadIdx.x; i < n;
       i += (long long)gridDim.x * blockDim.x)
    y[i] = (bf16)x[i];
}

// W[k][cin][cout] fp32  ->  Wt[k][cout][cin] bf16   (B-operand friendly layout)
__global__ void wt_cast_kernel(const float* __restrict__ w, bf16* __restrict__ wt,
                               int K, int cin, int cout) {
  long long total = (long long)K * cin * cout;
  for (long long i = (long long)blockIdx.x * blockDim.x + threadIdx.x; i < total;
       i += (long long)gridDim.x * blockDim.x) {
    int co = (int)(i % cout);
    long long t = i / cout;
    int ci = (int)(t % cin);
    int k  = (int)(t / cin);
    wt[((long long)k * cout + co) * cin + ci] = (bf16)w[i];
  }
}

__global__ void copy_cols_kernel(const float* __restrict__ x, long long n, int cs,
                                 float* __restrict__ y, int cd, int off) {
  long long total = n * cs;
  for (long long i = (long long)blockIdx.x * blockDim.x + threadIdx.x; i < total;
       i += (long long)gridDim.x * blockDim.x) {
    long long r = i / cs; int c = (int)(i % cs);
    y[r * cd + off + c] = x[i];
  }
}

// ---------------------------------------------------------------------------
// Core: gather -> WMMA(bf16, fp32 acc) -> atomic scatter.
// One wave computes a 16(map rows) x 32(cout) tile: the gathered A operand is
// loaded ONCE per K-step and fed to TWO v_wmma_f32_16x16x32_bf16 issues
// (doubles WMMA per gathered byte; gather traffic is the roofline limiter).
// A layout (16-bit A 16x32): lane m=L&15; elems 0-7 = k0+8*half.., 8-15 = +16.
// B layout mirrors with n=L&15 over the pre-transposed Wt rows.
// ---------------------------------------------------------------------------
__global__ void __launch_bounds__(32)
spconv_wmma_kernel(const bf16* __restrict__ X, const bf16* __restrict__ Wt,
                   float* __restrict__ Y, ConvMaps m, int cin, int cout) {
  const int off   = blockIdx.z;
  const int L     = m.len[off];
  const int tbase = blockIdx.x * 16;
  if (tbase >= L) return;                       // wave-uniform exit (EXEC stays full)
  const int lane = threadIdx.x;
  const int half = lane >> 4;
  const int l16  = lane & 15;
  const int* ii = m.ii[off];
  const int* oi = m.oi[off];

  int am   = tbase + l16;
  int arow = (am < L) ? (ii ? ii[am] : am) : 0; // clamp: extra rows masked on scatter
  const bf16* __restrict__ xrow = X + (size_t)arow * cin;
  const size_t wbase = ((size_t)off * cout + (size_t)(blockIdx.y * 32)) * cin;
  const bf16* __restrict__ wrow0 = Wt + wbase + (size_t)l16 * cin;        // cols 0..15
  const bf16* __restrict__ wrow1 = Wt + wbase + (size_t)(16 + l16) * cin; // cols 16..31

  v8f acc0 = {};
  v8f acc1 = {};
  union { v16bf v; int4 q[2]; } A, B0, B1;
  const int kh = 8 * half;
  for (int k0 = 0; k0 < cin; k0 += 32) {
    A.q[0]  = *(const int4*)(xrow  + k0 + kh);
    A.q[1]  = *(const int4*)(xrow  + k0 + kh + 16);
    B0.q[0] = *(const int4*)(wrow0 + k0 + kh);
    B0.q[1] = *(const int4*)(wrow0 + k0 + kh + 16);
    B1.q[0] = *(const int4*)(wrow1 + k0 + kh);
    B1.q[1] = *(const int4*)(wrow1 + k0 + kh + 16);
    __builtin_prefetch(xrow + k0 + 32, 0, 0);   // global_prefetch_b8: next K tile
    acc0 = __builtin_amdgcn_wmma_f32_16x16x32_bf16(
        false, A.v, false, B0.v, (short)0, acc0, false, false);
    acc1 = __builtin_amdgcn_wmma_f32_16x16x32_bf16(
        false, A.v, false, B1.v, (short)0, acc1, false, false);
  }

  const int col0 = blockIdx.y * 32 + l16;       // D: lane -> n, VGPR r -> m = r+8*half
#pragma unroll
  for (int r = 0; r < 8; ++r) {
    int mrow = tbase + r + 8 * half;
    if (mrow < L) {
      int orow = oi ? oi[mrow] : mrow;
      float* yrow = Y + (size_t)orow * cout;
      atomicAdd(yrow + col0,      acc0[r]);
      atomicAdd(yrow + col0 + 16, acc1[r]);
    }
  }
}

// Stem conv: cin=3 (not WMMA-shaped) -> scalar MACs, bandwidth bound anyway.
__global__ void spconv_c3_kernel(const float* __restrict__ X, const float* __restrict__ W,
                                 const int* __restrict__ ii, const int* __restrict__ oi,
                                 int L, int cout, float* __restrict__ Y) {
  long long total = (long long)L * cout;
  for (long long i = (long long)blockIdx.x * blockDim.x + threadIdx.x; i < total;
       i += (long long)gridDim.x * blockDim.x) {
    int co = (int)(i % cout);
    int e  = (int)(i / cout);
    const float* xr = X + (size_t)ii[e] * 3;
    float s = xr[0] * W[co] + xr[1] * W[cout + co] + xr[2] * W[2 * cout + co];
    atomicAdd(Y + (size_t)oi[e] * cout + co, s);
  }
}

// BN: pass 1 — per-channel sum & sumsq via LDS ds_add_f32 then global atomics.
__global__ void bn_reduce_kernel(const float* __restrict__ x, long long n, int c,
                                 float* __restrict__ stats) {
  extern __shared__ float sh[];                 // 2*c floats
  for (int i = threadIdx.x; i < 2 * c; i += blockDim.x) sh[i] = 0.0f;
  __syncthreads();
  long long total = n * c;
  for (long long i = (long long)blockIdx.x * blockDim.x + threadIdx.x; i < total;
       i += (long long)gridDim.x * blockDim.x) {
    float v = x[i]; int ch = (int)(i % c);
    atomicAdd(&sh[ch], v);
    atomicAdd(&sh[c + ch], v * v);
  }
  __syncthreads();
  for (int i = threadIdx.x; i < 2 * c; i += blockDim.x) atomicAdd(&stats[i], sh[i]);
}

// BN: pass 2 — normalize, optional residual add, optional ReLU.
__global__ void bn_apply_kernel(const float* __restrict__ x, long long n, int c,
                                const float* __restrict__ stats,
                                const float* __restrict__ g, const float* __restrict__ b,
                                const float* __restrict__ res, float* __restrict__ y,
                                int relu) {
  float invn = 1.0f / (float)n;
  long long total = n * c;
  for (long long i = (long long)blockIdx.x * blockDim.x + threadIdx.x; i < total;
       i += (long long)gridDim.x * blockDim.x) {
    int ch = (int)(i % c);
    float mean = stats[ch] * invn;
    float var  = stats[c + ch] * invn - mean * mean;
    float v = (x[i] - mean) * rsqrtf(var + 1e-5f) * g[ch] + b[ch];
    if (res) v += res[i];
    if (relu) v = fmaxf(v, 0.0f);
    y[i] = v;
  }
}

__global__ void linear_kernel(const float* __restrict__ x, const float* __restrict__ w,
                              const float* __restrict__ bias, float* __restrict__ y,
                              long long n, int cin, int cout) {
  long long total = n * cout;
  for (long long i = (long long)blockIdx.x * blockDim.x + threadIdx.x; i < total;
       i += (long long)gridDim.x * blockDim.x) {
    int co = (int)(i % cout);
    long long r = i / cout;
    const float* xr = x + r * cin;
    float s = bias[co];
    for (int k = 0; k < cin; ++k) s += xr[k] * w[(size_t)k * cout + co];
    y[i] = s;
  }
}

// ---------------------------------------------------------------------------
// Host orchestration
// ---------------------------------------------------------------------------
struct MapSet { const int* ii[MAXOFF]; const int* oi[MAXOFF]; int len[MAXOFF]; int K; };
struct BNP  { const float* g; const float* b; };
struct BlkP { const float* w1; const float* w2; BNP bn1, bn2; const float* wd; BNP bnd; };

extern "C" void kernel_launch(void* const* d_in, const int* in_sizes, int n_in,
                              void* d_out, int out_size, void* d_ws, size_t ws_size,
                              hipStream_t stream) {
  (void)n_in; (void)out_size; (void)ws_size;
  int idx = 0;
  auto takeF = [&]() -> const float* { return (const float*)d_in[idx++]; };

  auto takeBlk = [&](BlkP& bl, bool down) {
    bl.w1 = takeF(); bl.w2 = takeF();
    bl.bn1.g = takeF(); bl.bn1.b = takeF();
    bl.bn2.g = takeF(); bl.bn2.b = takeF();
    if (down) { bl.wd = takeF(); bl.bnd.g = takeF(); bl.bnd.b = takeF(); }
    else bl.wd = nullptr;
  };

  // ---- inputs in setup_inputs() insertion order ----
  const float* feats = takeF();                       // (n0, 3)
  const float* w0 = takeF();  BNP bn0{takeF(), takeF()};
  const float* wd1 = takeF(); BNP bn1{takeF(), takeF()};
  BlkP blk1[2]; takeBlk(blk1[0], false); takeBlk(blk1[1], false);
  const float* wd2 = takeF(); BNP bn2{takeF(), takeF()};
  BlkP blk2[2]; takeBlk(blk2[0], true);  takeBlk(blk2[1], false);
  const float* wd3 = takeF(); BNP bn3{takeF(), takeF()};
  BlkP blk3[2]; takeBlk(blk3[0], true);  takeBlk(blk3[1], false);
  const float* wd4 = takeF(); BNP bn4{takeF(), takeF()};
  BlkP blk4[2]; takeBlk(blk4[0], true);  takeBlk(blk4[1], false);
  const float* wu4 = takeF(); BNP bntr4{takeF(), takeF()};
  BlkP blk5[2]; takeBlk(blk5[0], true);  takeBlk(blk5[1], false);
  const float* wu5 = takeF(); BNP bntr5{takeF(), takeF()};
  BlkP blk6[2]; takeBlk(blk6[0], true);  takeBlk(blk6[1], false);
  const float* wu6 = takeF(); BNP bntr6{takeF(), takeF()};
  BlkP blk7[2]; takeBlk(blk7[0], true);  takeBlk(blk7[1], false);
  const float* wu7 = takeF(); BNP bntr7{takeF(), takeF()};
  BlkP blk8[2]; takeBlk(blk8[0], true);  takeBlk(blk8[1], false);
  const float* wf  = takeF(); const float* bfin = takeF();

  // kmaps: coords (5), then 14 map lists of (ii, oi) pairs
  long long n[5];
  for (int i = 0; i < 5; ++i) { n[i] = in_sizes[idx] / 4; idx++; }
  const int* c0ii[125]; const int* c0oi[125]; int c0len[125];
  for (int k = 0; k < 125; ++k) {
    c0ii[k] = (const int*)d_in[idx]; c0len[k] = in_sizes[idx]; idx++;
    c0oi[k] = (const int*)d_in[idx]; idx++;
  }
  auto takeMaps = [&](MapSet& ms, int K) {
    ms.K = K;
    for (int k = 0; k < K; ++k) {
      ms.ii[k] = (const int*)d_in[idx]; ms.len[k] = in_sizes[idx]; idx++;
      ms.oi[k] = (const int*)d_in[idx]; idx++;
    }
  };
  MapSet d1m, b1m, d2m, b2m, d3m, b3m, d4m, b4m, u4m, u5m, u6m, u7m, b8m;
  takeMaps(d1m, 8);  takeMaps(b1m, 27);
  takeMaps(d2m, 8);  takeMaps(b2m, 27);
  takeMaps(d3m, 8);  takeMaps(b3m, 27);
  takeMaps(d4m, 8);  takeMaps(b4m, 27);
  takeMaps(u4m, 8);  takeMaps(u5m, 8); takeMaps(u6m, 8); takeMaps(u7m, 8);
  takeMaps(b8m, 27);

  // ---- workspace bump allocator (256B aligned) ----
  char* wsp = (char*)d_ws; size_t wsoff = 0;
  auto alloc = [&](size_t bytes) -> void* {
    void* p = wsp + wsoff;
    wsoff += (bytes + 255) & ~(size_t)255;
    return p;
  };
  auto allocF = [&](size_t count) -> float* { return (float*)alloc(count * 4); };

  auto cdiv = [](long long a, long long b) -> long long { return (a + b - 1) / b; };
  auto gs   = [&](long long total) -> dim3 {
    long long g = cdiv(total, 256); if (g > 4096) g = 4096; if (g < 1) g = 1;
    return dim3((unsigned)g);
  };

  auto zero = [&](float* p, long long cnt) {
    fillz_kernel<<<gs(cnt), 256, 0, stream>>>(p, cnt);
  };
  auto castb = [&](const float* x, long long cnt) -> bf16* {
    bf16* y = (bf16*)alloc((size_t)cnt * sizeof(bf16));
    cast_bf16_kernel<<<gs(cnt), 256, 0, stream>>>(x, y, cnt);
    return y;
  };
  auto prepW = [&](const float* w, int K, int cin, int cout) -> bf16* {
    long long cnt = (long long)K * cin * cout;
    bf16* wt = (bf16*)alloc((size_t)cnt * sizeof(bf16));
    wt_cast_kernel<<<gs(cnt), 256, 0, stream>>>(w, wt, K, cin, cout);
    return wt;
  };
  auto convW = [&](const bf16* X, const MapSet& ms, const float* w,
                   int cin, int cout, long long nOut) -> float* {
    bf16* wt = prepW(w, ms.K, cin, cout);
    float* y = allocF((size_t)nOut * cout);
    zero(y, nOut * cout);
    ConvMaps cm{};
    int maxL = 1;
    for (int k = 0; k < ms.K; ++k) {
      cm.ii[k] = ms.ii[k]; cm.oi[k] = ms.oi[k]; cm.len[k] = ms.len[k];
      if (ms.len[k] > maxL) maxL = ms.len[k];
    }
    dim3 grid((unsigned)cdiv(maxL, 16), (unsigned)(cout / 32), (unsigned)ms.K);
    spconv_wmma_kernel<<<grid, 32, 0, stream>>>(X, wt, y, cm, cin, cout);
    return y;
  };
  auto denseW = [&](const bf16* X, long long nrows, const float* w,
                    int cin, int cout) -> float* {
    bf16* wt = prepW(w, 1, cin, cout);
    float* y = allocF((size_t)nrows * cout);
    zero(y, nrows * cout);
    ConvMaps cm{};
    cm.ii[0] = nullptr; cm.oi[0] = nullptr; cm.len[0] = (int)nrows;
    dim3 grid((unsigned)cdiv(nrows, 16), (unsigned)(cout / 32), 1);
    spconv_wmma_kernel<<<grid, 32, 0, stream>>>(X, wt, y, cm, cin, cout);
    return y;
  };
  auto bn = [&](const float* x, long long nrows, int c, BNP p,
                const float* res, bool relu, float* ydst = nullptr) -> float* {
    float* stats = allocF(2 * (size_t)c);
    zero(stats, 2 * c);
    bn_reduce_kernel<<<gs(nrows * c), 256, 2 * c * sizeof(float), stream>>>(
        x, nrows, c, stats);
    float* y = ydst ? ydst : allocF((size_t)nrows * c);
    bn_apply_kernel<<<gs(nrows * c), 256, 0, stream>>>(
        x, nrows, c, stats, p.g, p.b, res, y, relu ? 1 : 0);
    return y;
  };
  auto concat2 = [&](const float* a, int ca, const float* b2, int cb,
                     long long nrows) -> float* {
    int cd = ca + cb;
    float* y = allocF((size_t)nrows * cd);
    copy_cols_kernel<<<gs(nrows * ca), 256, 0, stream>>>(a, nrows, ca, y, cd, 0);
    copy_cols_kernel<<<gs(nrows * cb), 256, 0, stream>>>(b2, nrows, cb, y, cd, ca);
    return y;
  };
  auto blockf = [&](float* x, const MapSet& ms, long long nrows, const BlkP& bp,
                    int cin, int cout) -> float* {
    bf16* xb = castb(x, nrows * cin);
    float* t1 = convW(xb, ms, bp.w1, cin, cout, nrows);
    float* o  = bn(t1, nrows, cout, bp.bn1, nullptr, true);
    bf16* ob  = castb(o, nrows * cout);
    float* t2 = convW(ob, ms, bp.w2, cout, cout, nrows);
    const float* r = x;
    if (bp.wd) {
      float* rd = denseW(xb, nrows, bp.wd, cin, cout);
      r = bn(rd, nrows, cout, bp.bnd, nullptr, false);
    }
    return bn(t2, nrows, cout, bp.bn2, r, true);
  };

  // ---------------- forward pass ----------------
  // Stem: 5^3 conv, cin=3 -> 32
  float* a0 = allocF((size_t)n[0] * 32);
  zero(a0, n[0] * 32);
  for (int k = 0; k < 125; ++k)
    spconv_c3_kernel<<<gs((long long)c0len[k] * 32), 256, 0, stream>>>(
        feats, w0 + (size_t)k * 3 * 32, c0ii[k], c0oi[k], c0len[k], 32, a0);
  float* p1 = bn(a0, n[0], 32, bn0, nullptr, true);      // out_p1 (n0,32)

  // Encoder
  float* x = bn(convW(castb(p1, n[0] * 32), d1m, wd1, 32, 32, n[1]),
                n[1], 32, bn1, nullptr, true);
  x = blockf(x, b1m, n[1], blk1[0], 32, 32);
  x = blockf(x, b1m, n[1], blk1[1], 32, 32);
  float* outb1 = x;                                       // (n1,32)

  x = bn(convW(castb(x, n[1] * 32), d2m, wd2, 32, 32, n[2]),
         n[2], 32, bn2, nullptr, true);
  x = blockf(x, b2m, n[2], blk2[0], 32, 64);
  x = blockf(x, b2m, n[2], blk2[1], 64, 64);
  float* outb2 = x;                                       // (n2,64)

  x = bn(convW(castb(x, n[2] * 64), d3m, wd3, 64, 64, n[3]),
         n[3], 64, bn3, nullptr, true);
  x = blockf(x, b3m, n[3], blk3[0], 64, 128);
  x = blockf(x, b3m, n[3], blk3[1], 128, 128);
  float* outb3 = x;                                       // (n3,128)

  x = bn(convW(castb(x, n[3] * 128), d4m, wd4, 128, 128, n[4]),
         n[4], 128, bn4, nullptr, true);
  x = blockf(x, b4m, n[4], blk4[0], 128, 256);
  x = blockf(x, b4m, n[4], blk4[1], 256, 256);            // (n4,256)

  // Decoder
  x = bn(convW(castb(x, n[4] * 256), u4m, wu4, 256, 256, n[3]),
         n[3], 256, bntr4, nullptr, true);
  x = concat2(x, 256, outb3, 128, n[3]);                  // 384
  x = blockf(x, b3m, n[3], blk5[0], 384, 256);
  x = blockf(x, b3m, n[3], blk5[1], 256, 256);

  x = bn(convW(castb(x, n[3] * 256), u5m, wu5, 256, 128, n[2]),
         n[2], 128, bntr5, nullptr, true);
  x = concat2(x, 128, outb2, 64, n[2]);                   // 192
  x = blockf(x, b2m, n[2], blk6[0], 192, 128);
  x = blockf(x, b2m, n[2], blk6[1], 128, 128);

  x = bn(convW(castb(x, n[2] * 128), u6m, wu6, 128, 96, n[1]),
         n[1], 96, bntr6, nullptr, true);
  x = concat2(x, 96, outb1, 32, n[1]);                    // 128
  x = blockf(x, b1m, n[1], blk7[0], 128, 96);
  x = blockf(x, b1m, n[1], blk7[1], 96, 96);

  x = bn(convW(castb(x, n[1] * 96), u7m, wu7, 96, 96, n[0]),
         n[0], 96, bntr7, nullptr, true);
  x = concat2(x, 96, p1, 32, n[0]);                       // 128
  x = blockf(x, b8m, n[0], blk8[0], 128, 96);
  x = blockf(x, b8m, n[0], blk8[1], 96, 96);

  // Final classifier: (n0,96) @ (96,20) + bias -> d_out
  linear_kernel<<<gs(n[0] * 20), 256, 0, stream>>>(
      x, wf, bfin, (float*)d_out, n[0], 96, 20);
}
